// PerspectiveConv2d_40432822124900
// MI455X (gfx1250) — compile-verified, matrix operand order checked by reading, not run on hardware
//
#include <hip/hip_runtime.h>

typedef __attribute__((ext_vector_type(16))) __bf16 v16bf;
typedef __attribute__((ext_vector_type(8)))  float  v8f;

#define Bn 8
#define Cc 512
#define Hh 18
#define Ww 80
#define Oo 512
#define HW (Hh*Ww)            // 1440
#define CK (Cc*9)             // 4608
#define MTOT (Bn*HW)          // 11520
#define BLOCK_M 32
#define KC 64
#define NCHUNK (CK/KC)        // 72
#define NK32 (CK/32)          // 144 (16x16x32 k-steps)
#define NTILES (Oo/16)        // 32
#define LDK (KC+8)            // 72 elems -> 144B rows (16B aligned)
#define WSWZ_BYTES ((size_t)Oo*CK*2)   // 4,718,592 B of swizzled bf16 weights

union FragPack { uint4 q[2]; v16bf v; };

__device__ __forceinline__ unsigned short f2bf(float f) {
  unsigned u = __float_as_uint(f);
  u += 0x7FFFu + ((u >> 16) & 1u);
  return (unsigned short)(u >> 16);
}

// ---------------------------------------------------------------------------
// Pre-pass: weight (O, C*9) f32 -> bf16, K permuted to K' = t*512 + c, and
// swizzled so each WMMA B-fragment (16 N x 32 K') is stored in lane order:
// piece index = ((nt*NK32 + k32)*2 + p)*32 + lane, 16 bytes per piece.
// Lane l holds N = nt*16 + (l&15); piece p covers K'rel = p*16 + (l>>4)*8 + j.
// ---------------------------------------------------------------------------
__global__ __launch_bounds__(256)
void weight_swizzle(const float* __restrict__ wgt, unsigned short* __restrict__ dst)
{
  int flat = blockIdx.x * 256 + threadIdx.x;     // 294912 pieces
  int l    = flat & 31;
  int p    = (flat >> 5) & 1;
  int rest = flat >> 6;                          // nt*NK32 + kc
  int kc   = rest % NK32;
  int nt   = rest / NK32;
  int n    = nt * 16 + (l & 15);
  int k0   = kc * 32 + p * 16 + (l >> 4) * 8;    // 8 consecutive K'
  unsigned pk[4];
#pragma unroll
  for (int h = 0; h < 4; ++h) {
    int Kp0 = k0 + 2 * h;
    int t0 = Kp0 >> 9,       c0 = Kp0 & 511;
    int t1 = (Kp0 + 1) >> 9, c1 = (Kp0 + 1) & 511;
    unsigned short a = f2bf(wgt[n * CK + c0 * 9 + t0]);
    unsigned short b = f2bf(wgt[n * CK + c1 * 9 + t1]);
    pk[h] = (unsigned)a | ((unsigned)b << 16);
  }
  uint4 q = make_uint4(pk[0], pk[1], pk[2], pk[3]);
  *(uint4*)&dst[(size_t)flat * 8] = q;
}

// ---------------------------------------------------------------------------
// Main fused deform-im2col + WMMA GEMM (fast path, swizzled bf16 weights)
// ---------------------------------------------------------------------------
__global__ __launch_bounds__(256)
void deform_conv_wmma_fast(const float* __restrict__ feat,
                           const float* __restrict__ offs,
                           const unsigned short* __restrict__ bwz,
                           float* __restrict__ out)
{
  __shared__ __align__(16) unsigned short Abuf[2][BLOCK_M][LDK];  // 2x4.5KB
  __shared__ __align__(16) int   sOff[BLOCK_M][9][4];
  __shared__ __align__(16) float sWgt[BLOCK_M][9][4];

  const int tid   = threadIdx.x;
  const int lane  = tid & 31;
  const int wave  = tid >> 5;
  const int l16   = lane & 15;
  const int lhalf = lane >> 4;
  const int m_base = blockIdx.x * BLOCK_M;

  // gather assignment: lanes run along m (spatially adjacent -> coalescable
  // gather addresses); each thread covers 8 consecutive K' in its row.
  const int gm  = tid & 31;
  const int kk0 = (tid >> 5) * 8;

  // ---- precompute bilinear tap descriptors per (m-row, tap) ----
  for (int idx = tid; idx < BLOCK_M * 9; idx += 256) {
    int m  = idx / 9, t = idx - 9 * m;
    int mg = m_base + m;
    int b  = mg / HW;
    int rem = mg - b * HW;
    int h  = rem / Ww;
    int w  = rem - h * Ww;
    float dy = offs[((b * 18 + 2 * t    ) * Hh + h) * Ww + w];
    float dx = offs[((b * 18 + 2 * t + 1) * Hh + h) * Ww + w];
    int   kh = t / 3, kw = t - 3 * kh;
    float py = dy + (float)(h - 1 + kh);
    float px = dx + (float)(w - 1 + kw);
    float fy = floorf(py), fx = floorf(px);
    int   y0 = (int)fy,    x0 = (int)fx;
    float ly = py - fy,    lx = px - fx;
    float wy[2] = {1.0f - ly, ly};
    float wx[2] = {1.0f - lx, lx};
    int bbase = b * Cc * HW;
#pragma unroll
    for (int j = 0; j < 4; ++j) {
      int yi = y0 + (j >> 1);
      int xi = x0 + (j & 1);
      bool ok = (yi >= 0) && (yi < Hh) && (xi >= 0) && (xi < Ww);
      int yc = min(max(yi, 0), Hh - 1);
      int xc = min(max(xi, 0), Ww - 1);
      sOff[m][t][j] = bbase + yc * Ww + xc;
      sWgt[m][t][j] = ok ? wy[j >> 1] * wx[j & 1] : 0.0f;
    }
  }
  __syncthreads();

  v8f acc[2][4];
#pragma unroll
  for (int i = 0; i < 2; ++i)
#pragma unroll
    for (int j = 0; j < 4; ++j) { v8f z = {}; acc[i][j] = z; }

  // ---- prologue: fill A buffer 0 (chunk 0: t = 0, c = kk) ----
  {
    int4   o4 = *(const int4*)  &sOff[gm][0][0];
    float4 w4 = *(const float4*)&sWgt[gm][0][0];
    int cb = kk0 * HW;
    unsigned pk[4];
#pragma unroll
    for (int h = 0; h < 4; ++h) {
      float v0 = w4.x * feat[o4.x + cb] + w4.y * feat[o4.y + cb]
               + w4.z * feat[o4.z + cb] + w4.w * feat[o4.w + cb];
      cb += HW;
      float v1 = w4.x * feat[o4.x + cb] + w4.y * feat[o4.y + cb]
               + w4.z * feat[o4.z + cb] + w4.w * feat[o4.w + cb];
      cb += HW;
      pk[h] = (unsigned)f2bf(v0) | ((unsigned)f2bf(v1) << 16);
    }
    *(uint4*)&Abuf[0][gm][kk0] = make_uint4(pk[0], pk[1], pk[2], pk[3]);
  }
  __syncthreads();

  const uint4* bswq = (const uint4*)bwz;   // 16B pieces, lane-indexed

  for (int chunk = 0; chunk < NCHUNK; ++chunk) {
    const int cur = chunk & 1;

    // ---- consume: fragments + 16 WMMA ----
#pragma unroll
    for (int ks = 0; ks < 2; ++ks) {
      const int koff = ks * 32;
      FragPack a[2];
#pragma unroll
      for (int ms = 0; ms < 2; ++ms) {
        const uint4* row = (const uint4*)&Abuf[cur][ms * 16 + l16][koff];
        a[ms].q[0] = row[lhalf];
        a[ms].q[1] = row[2 + lhalf];
      }
      const int k32 = chunk * 2 + ks;
#pragma unroll
      for (int ns = 0; ns < 4; ++ns) {
        int nt = wave * 4 + ns;
        int pb = ((nt * NK32 + k32) * 2) * 32 + lane;
        FragPack bf;
        bf.q[0] = bswq[pb];
        bf.q[1] = bswq[pb + 32];
#pragma unroll
        for (int ms = 0; ms < 2; ++ms) {
          acc[ms][ns] = __builtin_amdgcn_wmma_f32_16x16x32_bf16(
              false, a[ms].v, false, bf.v, (short)0, acc[ms][ns], false, false);
        }
      }
    }

    // ---- fill next chunk's A tile (overlaps with WMMA above) ----
    if (chunk + 1 < NCHUNK) {
      const int kb = (chunk + 1) * KC;   // K' base
      const int t  = kb >> 9;            // constant tap for whole chunk
      const int c0 = (kb & 511) + kk0;
      int4   o4 = *(const int4*)  &sOff[gm][t][0];
      float4 w4 = *(const float4*)&sWgt[gm][t][0];
      int cb = c0 * HW;
      unsigned pk[4];
#pragma unroll
      for (int h = 0; h < 4; ++h) {
        float v0 = w4.x * feat[o4.x + cb] + w4.y * feat[o4.y + cb]
                 + w4.z * feat[o4.z + cb] + w4.w * feat[o4.w + cb];
        cb += HW;
        float v1 = w4.x * feat[o4.x + cb] + w4.y * feat[o4.y + cb]
                 + w4.z * feat[o4.z + cb] + w4.w * feat[o4.w + cb];
        cb += HW;
        pk[h] = (unsigned)f2bf(v0) | ((unsigned)f2bf(v1) << 16);
      }
      *(uint4*)&Abuf[cur ^ 1][gm][kk0] = make_uint4(pk[0], pk[1], pk[2], pk[3]);
    }
    __syncthreads();
  }

  // ---- store: C/D layout VGPR r -> M = r + 8*lhalf, N = lane&15 ----
#pragma unroll
  for (int ms = 0; ms < 2; ++ms) {
    int mg0 = m_base + ms * 16 + lhalf * 8;
#pragma unroll
    for (int r = 0; r < 8; ++r) {
      int mg = mg0 + r;
      int b  = mg / HW;
      int ob = mg + b * (Oo - 1) * HW;
#pragma unroll
      for (int ns = 0; ns < 4; ++ns) {
        int n = wave * 64 + ns * 16 + l16;
        out[ob + n * HW] = acc[ms][ns][r];
      }
    }
  }
}

// ---------------------------------------------------------------------------
// Fallback (no workspace needed)
// ---------------------------------------------------------------------------
__global__ __launch_bounds__(256)
void deform_conv_wmma_fb(const float* __restrict__ feat,
                         const float* __restrict__ offs,
                         const float* __restrict__ wgt,
                         float* __restrict__ out)
{
  __shared__ __align__(16) unsigned short Ash[BLOCK_M][40];
  __shared__ __align__(16) unsigned short Bsh[Oo][40];
  __shared__ int   sOff[BLOCK_M][9][4];
  __shared__ float sWgt[BLOCK_M][9][4];

  const int tid   = threadIdx.x;
  const int lane  = tid & 31;
  const int wave  = tid >> 5;
  const int l16   = lane & 15;
  const int lhalf = lane >> 4;
  const int m_base = blockIdx.x * BLOCK_M;

  for (int idx = tid; idx < BLOCK_M * 9; idx += 256) {
    int m  = idx / 9, t = idx - 9 * m;
    int mg = m_base + m;
    int b  = mg / HW;
    int rem = mg - b * HW;
    int h  = rem / Ww;
    int w  = rem - h * Ww;
    float dy = offs[((b * 18 + 2 * t    ) * Hh + h) * Ww + w];
    float dx = offs[((b * 18 + 2 * t + 1) * Hh + h) * Ww + w];
    int   kh = t / 3, kw = t - 3 * kh;
    float py = dy + (float)(h - 1 + kh);
    float px = dx + (float)(w - 1 + kw);
    float fy = floorf(py), fx = floorf(px);
    int   y0 = (int)fy,    x0 = (int)fx;
    float ly = py - fy,    lx = px - fx;
    float wy[2] = {1.0f - ly, ly};
    float wx[2] = {1.0f - lx, lx};
    int bbase = b * Cc * HW;
#pragma unroll
    for (int j = 0; j < 4; ++j) {
      int yi = y0 + (j >> 1);
      int xi = x0 + (j & 1);
      bool ok = (yi >= 0) && (yi < Hh) && (xi >= 0) && (xi < Ww);
      int yc = min(max(yi, 0), Hh - 1);
      int xc = min(max(xi, 0), Ww - 1);
      sOff[m][t][j] = bbase + yc * Ww + xc;
      sWgt[m][t][j] = ok ? wy[j >> 1] * wx[j & 1] : 0.0f;
    }
  }

  v8f acc[2][4];
#pragma unroll
  for (int i = 0; i < 2; ++i)
#pragma unroll
    for (int j = 0; j < 4; ++j) { v8f z = {}; acc[i][j] = z; }

  for (int chunk = 0; chunk < CK / 32; ++chunk) {
    const int kbase = chunk * 32;
    __syncthreads();
#pragma unroll
    for (int i = 0; i < 4; ++i) {
      int e  = i * 256 + tid;
      int m  = e >> 5;
      int kk = e & 31;
      int K  = kbase + kk;
      int c  = K / 9;
      int t  = K - 9 * c;
      const int*   o4 = sOff[m][t];
      const float* w4 = sWgt[m][t];
      int cb = c * HW;
      float v = w4[0] * feat[o4[0] + cb] + w4[1] * feat[o4[1] + cb]
              + w4[2] * feat[o4[2] + cb] + w4[3] * feat[o4[3] + cb];
      Ash[m][kk] = f2bf(v);
    }
#pragma unroll
    for (int j = 0; j < 32; ++j) {
      int s  = j * 256 + tid;
      int n  = s >> 4;
      int k2 = (s & 15) << 1;
      float2 wv = *(const float2*)&wgt[n * CK + kbase + k2];
      unsigned pk = (unsigned)f2bf(wv.x) | ((unsigned)f2bf(wv.y) << 16);
      *(unsigned*)&Bsh[n][k2] = pk;
    }
    __syncthreads();

    FragPack a[2];
#pragma unroll
    for (int ms = 0; ms < 2; ++ms) {
      const uint4* row = (const uint4*)&Ash[ms * 16 + l16][0];
      a[ms].q[0] = row[lhalf];
      a[ms].q[1] = row[2 + lhalf];
    }
#pragma unroll
    for (int ns = 0; ns < 4; ++ns) {
      FragPack bfr;
      const uint4* row = (const uint4*)&Bsh[wave * 64 + ns * 16 + l16][0];
      bfr.q[0] = row[lhalf];
      bfr.q[1] = row[2 + lhalf];
#pragma unroll
      for (int ms = 0; ms < 2; ++ms) {
        acc[ms][ns] = __builtin_amdgcn_wmma_f32_16x16x32_bf16(
            false, a[ms].v, false, bfr.v, (short)0, acc[ms][ns], false, false);
      }
    }
  }

#pragma unroll
  for (int ms = 0; ms < 2; ++ms) {
    int mg0 = m_base + ms * 16 + lhalf * 8;
#pragma unroll
    for (int r = 0; r < 8; ++r) {
      int mg = mg0 + r;
      int b  = mg / HW;
      int ob = mg + b * (Oo - 1) * HW;
#pragma unroll
      for (int ns = 0; ns < 4; ++ns) {
        int n = wave * 64 + ns * 16 + l16;
        out[ob + n * HW] = acc[ms][ns][r];
      }
    }
  }
}

extern "C" void kernel_launch(void* const* d_in, const int* in_sizes, int n_in,
                              void* d_out, int out_size, void* d_ws, size_t ws_size,
                              hipStream_t stream) {
  (void)in_sizes; (void)n_in; (void)out_size;
  const float* feat = (const float*)d_in[0];   // (8, 512, 18, 80)
  const float* offs = (const float*)d_in[1];   // (8, 18, 18, 80)
  const float* wgt  = (const float*)d_in[2];   // (512, 512, 3, 3)
  float* out = (float*)d_out;                  // (8, 512, 18, 80)

  if (ws_size >= WSWZ_BYTES) {
    unsigned short* bwz = (unsigned short*)d_ws;
    weight_swizzle<<<dim3((NTILES * NK32 * 2 * 32) / 256), 256, 0, stream>>>(wgt, bwz);
    deform_conv_wmma_fast<<<dim3(MTOT / BLOCK_M), 256, 0, stream>>>(feat, offs, bwz, out);
  } else {
    deform_conv_wmma_fb<<<dim3(MTOT / BLOCK_M), 256, 0, stream>>>(feat, offs, wgt, out);
  }
}
